// FA_LocalPointnet_65231963292427
// MI455X (gfx1250) — matrix operand order repaired and checked by reading, not compiled, not run on hardware
//
#include <hip/hip_runtime.h>
#include <hip/hip_bf16.h>

typedef _Float16 half_t;
typedef _Float16 v4h  __attribute__((ext_vector_type(4)));
typedef _Float16 v8h  __attribute__((ext_vector_type(8)));
typedef _Float16 v16h __attribute__((ext_vector_type(16)));
typedef float    v8f  __attribute__((ext_vector_type(8)));

#define KSZ        20
#define NPTS       1024
#define NBATCH     2
#define MROWS      16384      // B*8*N
#define NEG_SLOPE  0.2f
#define BN_EPS     1e-5f
#define CCW        5248       // concat width
#define LDSTRIDE   40         // LDS col stride in halves (80B, bank-spread)

// OPS8[o][i] sign: bit (2-i) of o -> -1
__device__ __forceinline__ float sgn_o(int o, int i) {
    return ((o >> (2 - i)) & 1) ? -1.0f : 1.0f;
}

__device__ __forceinline__ v16h frag16(const half_t* lo, const half_t* hi) {
    v8h a = *(const v8h*)lo;
    v8h b = *(const v8h*)hi;
    return __builtin_shufflevector(a, b, 0, 1, 2, 3, 4, 5, 6, 7,
                                   8, 9, 10, 11, 12, 13, 14, 15);
}

// ---------------- 3x3 symmetric eigen (ascending), columns = eigenvectors ----
__device__ void eigvec3(float a00, float a01, float a02, float a11, float a12,
                        float a22, float l, float v[3]) {
    float r0x = a00 - l, r0y = a01, r0z = a02;
    float r1x = a01, r1y = a11 - l, r1z = a12;
    float r2x = a02, r2y = a12, r2z = a22 - l;
    float c0x = r0y * r1z - r0z * r1y, c0y = r0z * r1x - r0x * r1z, c0z = r0x * r1y - r0y * r1x;
    float c1x = r0y * r2z - r0z * r2y, c1y = r0z * r2x - r0x * r2z, c1z = r0x * r2y - r0y * r2x;
    float c2x = r1y * r2z - r1z * r2y, c2y = r1z * r2x - r1x * r2z, c2z = r1x * r2y - r1y * r2x;
    float n0 = c0x * c0x + c0y * c0y + c0z * c0z;
    float n1 = c1x * c1x + c1y * c1y + c1z * c1z;
    float n2 = c2x * c2x + c2y * c2y + c2z * c2z;
    float bx = c0x, by = c0y, bz = c0z, bn = n0;
    if (n1 > bn) { bx = c1x; by = c1y; bz = c1z; bn = n1; }
    if (n2 > bn) { bx = c2x; by = c2y; bz = c2z; bn = n2; }
    float inv = rsqrtf(fmaxf(bn, 1e-30f));
    v[0] = bx * inv; v[1] = by * inv; v[2] = bz * inv;
}

__device__ void eigh3(float a00, float a01, float a02, float a11, float a12,
                      float a22, float V[9]) {
    float p1 = a01 * a01 + a02 * a02 + a12 * a12;
    if (p1 < 1e-18f) {  // (near-)diagonal: permutation sorted ascending
        float d[3] = {a00, a11, a22};
        int idx[3] = {0, 1, 2};
        for (int a = 0; a < 2; ++a)
            for (int bb = 0; bb < 2 - a; ++bb)
                if (d[bb] > d[bb + 1]) {
                    float td = d[bb]; d[bb] = d[bb + 1]; d[bb + 1] = td;
                    int ti = idx[bb]; idx[bb] = idx[bb + 1]; idx[bb + 1] = ti;
                }
        for (int i = 0; i < 9; ++i) V[i] = 0.f;
        for (int c = 0; c < 3; ++c) V[idx[c] * 3 + c] = 1.f;
        return;
    }
    float q = (a00 + a11 + a22) * (1.0f / 3.0f);
    float b00 = a00 - q, b11 = a11 - q, b22 = a22 - q;
    float p2 = b00 * b00 + b11 * b11 + b22 * b22 + 2.f * p1;
    float p = sqrtf(p2 * (1.0f / 6.0f));
    float ip = 1.0f / p;
    float c00 = b00 * ip, c01 = a01 * ip, c02 = a02 * ip;
    float c11 = b11 * ip, c12 = a12 * ip, c22 = b22 * ip;
    float det = c00 * (c11 * c22 - c12 * c12) - c01 * (c01 * c22 - c12 * c02) +
                c02 * (c01 * c12 - c11 * c02);
    float r = fminf(1.f, fmaxf(-1.f, 0.5f * det));
    float phi = acosf(r) * (1.0f / 3.0f);
    float eA = q + 2.f * p * cosf(phi);                        // largest
    float eC = q + 2.f * p * cosf(phi + 2.0943951023931953f);  // smallest
    float v0[3], v2[3];
    eigvec3(a00, a01, a02, a11, a12, a22, eC, v0);
    eigvec3(a00, a01, a02, a11, a12, a22, eA, v2);
    float v1[3] = {v2[1] * v0[2] - v2[2] * v0[1],
                   v2[2] * v0[0] - v2[0] * v0[2],
                   v2[0] * v0[1] - v2[1] * v0[0]};
    float n1 = v1[0] * v1[0] + v1[1] * v1[1] + v1[2] * v1[2];
    float inv = rsqrtf(fmaxf(n1, 1e-30f));
    v1[0] *= inv; v1[1] *= inv; v1[2] *= inv;
    for (int i = 0; i < 3; ++i) {
        V[i * 3 + 0] = v0[i];  // col 0 = smallest eigval
        V[i * 3 + 1] = v1[i];
        V[i * 3 + 2] = v2[i];  // col 2 = largest
    }
}

// ---------------- weight prep: W[k][n] fp32 -> Wt[n][k] fp16, zero padded ----
__global__ __launch_bounds__(256) void prep_w_kernel(const float* __restrict__ W,
        int din, int dout, half_t* __restrict__ Wt, int Kpad, int Npad) {
    int idx = blockIdx.x * 256 + threadIdx.x;
    if (idx >= Npad * Kpad) return;
    int n = idx / Kpad, k = idx % Kpad;
    float v = (k < din && n < dout) ? W[(size_t)k * dout + n] : 0.f;
    Wt[idx] = (half_t)v;
}

// ---------------- WMMA GEMM: Y[M][Npad] = A[M][K] * Wt^T + bias (fp32 out) ---
// block = 8 waves, tile 256 rows x 64 cols; wave tile 32x64 (8 accumulators).
// B tile (32K x 64N) double-buffered in LDS, register-staged prefetch,
// one barrier per K-step; 4 B fragments gathered up-front so the 8 WMMAs
// issue back-to-back behind a single dscnt wait.
__global__ __launch_bounds__(256) void gemm_kernel(
        const half_t* __restrict__ A, int lda,
        const half_t* __restrict__ Wt, int ldw,
        const float* __restrict__ bias, int dout,
        float* __restrict__ Y, int ldy, int K, int Npad) {
    __shared__ half_t sB[2][64 * LDSTRIDE];  // 2 x 5120B
    int tid = threadIdx.x;
    int wave = tid >> 5, lane = tid & 31;
    int half_id = lane >> 4, ln = lane & 15;
    int rowBase = blockIdx.y * 256 + wave * 32;
    int c0 = blockIdx.x * 64;

    v8f acc[2][4] = {};

    // cooperative B-tile load: 256 threads x 8 halves (one b128 each)
    int colL = tid >> 2;         // 0..63
    int kkL  = (tid & 3) * 8;    // 0,8,16,24
    const half_t* wsrc = Wt + (size_t)(c0 + colL) * ldw + kkL;
    half_t* wdst0 = &sB[0][colL * LDSTRIDE + kkL];
    half_t* wdst1 = &sB[1][colL * LDSTRIDE + kkL];

    const half_t* abase0 = A + (size_t)(rowBase + ln) * lda + half_id * 8;
    const half_t* abase1 = A + (size_t)(rowBase + 16 + ln) * lda + half_id * 8;

    // preload tile 0 into buffer 0
    v8h wreg = *(const v8h*)(wsrc);
    *(v8h*)wdst0 = wreg;
    __syncthreads();

    const int nsteps = K >> 5;
    for (int s = 0; s < nsteps; ++s) {
        int k0 = s << 5;
        const half_t* sb = sB[s & 1];
        bool more = (s + 1) < nsteps;
        if (more) wreg = *(const v8h*)(wsrc + k0 + 32);  // prefetch next tile

        v16h a0 = frag16(abase0 + k0, abase0 + k0 + 16);
        v16h a1 = frag16(abase1 + k0, abase1 + k0 + 16);

        v16h bf[4];
#pragma unroll
        for (int cb = 0; cb < 4; ++cb) {
            const half_t* bp = &sb[(cb * 16 + ln) * LDSTRIDE + half_id * 8];
            bf[cb] = frag16(bp, bp + 16);
        }
#pragma unroll
        for (int cb = 0; cb < 4; ++cb) {
            acc[0][cb] = __builtin_amdgcn_wmma_f32_16x16x32_f16(
                false, a0, false, bf[cb], (short)0, acc[0][cb], false, false);
            acc[1][cb] = __builtin_amdgcn_wmma_f32_16x16x32_f16(
                false, a1, false, bf[cb], (short)0, acc[1][cb], false, false);
        }
        if (more) *(v8h*)((s & 1) ? wdst0 : wdst1) = wreg;
        __syncthreads();
    }
#pragma unroll
    for (int rg = 0; rg < 2; ++rg)
#pragma unroll
        for (int cb = 0; cb < 4; ++cb)
#pragma unroll
            for (int v = 0; v < 8; ++v) {
                int row = rowBase + rg * 16 + v + half_id * 8;
                int col = c0 + cb * 16 + ln;
                float bb = (col < dout) ? bias[col] : 0.f;
                Y[(size_t)row * ldy + col] = acc[rg][cb][v] + bb;
            }
}

// ---------------- per-channel sum / sumsq (deterministic tree) --------------
__global__ __launch_bounds__(256) void stats_kernel(const float* __restrict__ Y,
        int ldy, float* __restrict__ stats) {
    int c = blockIdx.x, t = threadIdx.x;
    float s = 0.f, s2 = 0.f;
    for (int r = t; r < MROWS; r += 256) {
        float y = Y[(size_t)r * ldy + c];
        s += y; s2 += y * y;
    }
    __shared__ float sh[512];
    sh[t] = s; sh[256 + t] = s2;
    __syncthreads();
    for (int w = 128; w > 0; w >>= 1) {
        if (t < w) { sh[t] += sh[t + w]; sh[256 + t] += sh[256 + t + w]; }
        __syncthreads();
    }
    if (t == 0) { stats[2 * c] = sh[0]; stats[2 * c + 1] = sh[256]; }
}

// ---------------- BN apply (+ optional leaky), write fp16 and/or fp32 -------
__global__ __launch_bounds__(256) void bn_apply_kernel(const float* __restrict__ Y,
        int ldy, const float* __restrict__ stats, const float* __restrict__ gam,
        const float* __restrict__ bet, int dout, int Npad,
        half_t* __restrict__ dst16, int ld16, float* __restrict__ dst32, int ld32,
        float invM, int act) {
    int idx = blockIdx.x * 256 + threadIdx.x;
    int r = idx / Npad, c = idx % Npad;
    if (r >= MROWS) return;
    float v = 0.f;
    if (c < dout) {
        float y = Y[(size_t)r * ldy + c];
        float mean = stats[2 * c] * invM;
        float var = stats[2 * c + 1] * invM - mean * mean;
        float rs = rsqrtf(var + BN_EPS);
        v = gam[c] * (y - mean) * rs + bet[c];
        if (act) v = (v >= 0.f) ? v : NEG_SLOPE * v;
    }
    if (dst16) dst16[(size_t)r * ld16 + c] = (half_t)v;
    if (dst32) dst32[(size_t)r * ld32 + c] = v;
}

// ---------------- KNN + local frame + framed (fp16, 64 cols) ----------------
__global__ __launch_bounds__(256) void knn_frame_kernel(const float* __restrict__ P,
        half_t* __restrict__ framed, float* __restrict__ Vl) {
    __shared__ float sx[NPTS], sy[NPTS], sz[NPTS];
    int b = blockIdx.x >> 2;
    int nbase = (blockIdx.x & 3) << 8;
    for (int j = threadIdx.x; j < NPTS; j += 256) {
        sx[j] = P[(size_t)b * 3 * NPTS + j];
        sy[j] = P[(size_t)b * 3 * NPTS + NPTS + j];
        sz[j] = P[(size_t)b * 3 * NPTS + 2 * NPTS + j];
    }
    __syncthreads();
    int n = nbase + threadIdx.x;
    float px = sx[n], py = sy[n], pz = sz[n];
    float bd[KSZ]; int bi[KSZ]; int cnt = 0;
    for (int j = 0; j < NPTS; ++j) {
        float dx = sx[j] - px, dy = sy[j] - py, dz = sz[j] - pz;
        float d = dx * dx + dy * dy + dz * dz;
        if (cnt < KSZ) {
            int p = cnt++;
            while (p > 0 && bd[p - 1] > d) { bd[p] = bd[p - 1]; bi[p] = bi[p - 1]; --p; }
            bd[p] = d; bi[p] = j;
        } else if (d < bd[KSZ - 1]) {
            int p = KSZ - 1;
            while (p > 0 && bd[p - 1] > d) { bd[p] = bd[p - 1]; bi[p] = bi[p - 1]; --p; }
            bd[p] = d; bi[p] = j;
        }
    }
    float nx[KSZ], ny[KSZ], nz[KSZ];
    float cx = 0.f, cy = 0.f, cz = 0.f;
    for (int k = 0; k < KSZ; ++k) {
        nx[k] = sx[bi[k]]; ny[k] = sy[bi[k]]; nz[k] = sz[bi[k]];
        cx += nx[k]; cy += ny[k]; cz += nz[k];
    }
    cx *= (1.0f / KSZ); cy *= (1.0f / KSZ); cz *= (1.0f / KSZ);
    float a00 = 0, a01 = 0, a02 = 0, a11 = 0, a12 = 0, a22 = 0;
    for (int k = 0; k < KSZ; ++k) {
        float dx = nx[k] - cx, dy = ny[k] - cy, dz = nz[k] - cz;
        a00 += dx * dx; a01 += dx * dy; a02 += dx * dz;
        a11 += dy * dy; a12 += dy * dz; a22 += dz * dz;
    }
    float V[9];
    eigh3(a00, a01, a02, a11, a12, a22, V);
    size_t vb = ((size_t)b * NPTS + n) * 9;
    for (int t = 0; t < 9; ++t) Vl[vb + t] = V[t];
    for (int o = 0; o < 8; ++o) {
        float s0 = sgn_o(o, 0), s1 = sgn_o(o, 1), s2v = sgn_o(o, 2);
        size_t row = (((size_t)b * 8 + o) * NPTS + n) * 64;
        for (int k = 0; k < KSZ; ++k) {
            float dx = nx[k] - cx, dy = ny[k] - cy, dz = nz[k] - cz;
            float p0 = dx * V[0] + dy * V[3] + dz * V[6];
            float p1 = dx * V[1] + dy * V[4] + dz * V[7];
            float p2 = dx * V[2] + dy * V[5] + dz * V[8];
            framed[row + k * 3 + 0] = (half_t)(s0 * p0);
            framed[row + k * 3 + 1] = (half_t)(s1 * p1);
            framed[row + k * 3 + 2] = (half_t)(s2v * p2);
        }
        framed[row + 60] = (half_t)0.f; framed[row + 61] = (half_t)0.f;
        framed[row + 62] = (half_t)0.f; framed[row + 63] = (half_t)0.f;
    }
}

// ---------------- equiv_f from e3 output (fp32) ------------------------------
__global__ __launch_bounds__(256) void equivf_kernel(const float* __restrict__ h3,
        const float* __restrict__ Vl, float* __restrict__ eqf,
        float* __restrict__ outp) {
    int idx = blockIdx.x * 256 + threadIdx.x;  // over B*N*42
    if (idx >= NBATCH * NPTS * 42) return;
    int r = idx % 42; int bn = idx / 42;
    int n = bn % NPTS; int b = bn / NPTS;
    float g0 = 0.f, g1 = 0.f, g2 = 0.f;
    for (int o = 0; o < 8; ++o) {
        size_t base = (((size_t)b * 8 + o) * NPTS + n) * 128 + r * 3;
        g0 += sgn_o(o, 0) * h3[base + 0];
        g1 += sgn_o(o, 1) * h3[base + 1];
        g2 += sgn_o(o, 2) * h3[base + 2];
    }
    g0 *= 0.125f; g1 *= 0.125f; g2 *= 0.125f;
    const float* V = Vl + ((size_t)b * NPTS + n) * 9;
    for (int i = 0; i < 3; ++i) {
        float v = V[i * 3 + 0] * g0 + V[i * 3 + 1] * g1 + V[i * 3 + 2] * g2;
        size_t oidx = ((size_t)bn * 42 + r) * 3 + i;
        eqf[oidx] = v; outp[oidx] = v;
    }
}

// ---------------- global frame: mean + covariance + eigh ---------------------
__global__ __launch_bounds__(256) void gframe_kernel(const float* __restrict__ eqf,
        float* __restrict__ c2g, float* __restrict__ Vg) {
    int b = blockIdx.x, t = threadIdx.x;
    const float* X = eqf + (size_t)b * NPTS * 42 * 3;
    const int NPT = NPTS * 42;
    float acc[9] = {0, 0, 0, 0, 0, 0, 0, 0, 0};
    for (int i = t; i < NPT; i += 256) {
        float x = X[i * 3], y = X[i * 3 + 1], z = X[i * 3 + 2];
        acc[0] += x; acc[1] += y; acc[2] += z;
        acc[3] += x * x; acc[4] += x * y; acc[5] += x * z;
        acc[6] += y * y; acc[7] += y * z; acc[8] += z * z;
    }
    __shared__ float sh[9 * 256];
    for (int j = 0; j < 9; ++j) sh[j * 256 + t] = acc[j];
    __syncthreads();
    for (int w = 128; w > 0; w >>= 1) {
        if (t < w)
            for (int j = 0; j < 9; ++j) sh[j * 256 + t] += sh[j * 256 + t + w];
        __syncthreads();
    }
    if (t == 0) {
        float fn = (float)NPT;
        float cx = sh[0] / fn, cy = sh[256] / fn, cz = sh[512] / fn;
        float a00 = sh[3 * 256] - fn * cx * cx;
        float a01 = sh[4 * 256] - fn * cx * cy;
        float a02 = sh[5 * 256] - fn * cx * cz;
        float a11 = sh[6 * 256] - fn * cy * cy;
        float a12 = sh[7 * 256] - fn * cy * cz;
        float a22 = sh[8 * 256] - fn * cz * cz;
        float V[9];
        eigh3(a00, a01, a02, a11, a12, a22, V);
        c2g[b * 3 + 0] = cx; c2g[b * 3 + 1] = cy; c2g[b * 3 + 2] = cz;
        for (int j = 0; j < 9; ++j) Vg[b * 9 + j] = V[j];
    }
}

// ---------------- framed2 (fp16, 128 cols) -----------------------------------
__global__ __launch_bounds__(256) void framed2_kernel(const float* __restrict__ eqf,
        const float* __restrict__ c2g, const float* __restrict__ Vg,
        half_t* __restrict__ fr2) {
    int idx = blockIdx.x * 256 + threadIdx.x;  // over MROWS
    if (idx >= MROWS) return;
    int n = idx % NPTS; int bo = idx / NPTS; int o = bo % 8; int b = bo / 8;
    const float* V = Vg + b * 9;
    const float* c = c2g + b * 3;
    float s0 = sgn_o(o, 0), s1 = sgn_o(o, 1), s2 = sgn_o(o, 2);
    half_t* dst = fr2 + (size_t)idx * 128;
    const float* X = eqf + ((size_t)b * NPTS + n) * 126;
    for (int r = 0; r < 42; ++r) {
        float x = X[r * 3] - c[0], y = X[r * 3 + 1] - c[1], z = X[r * 3 + 2] - c[2];
        float p0 = V[0] * x + V[3] * y + V[6] * z;
        float p1 = V[1] * x + V[4] * y + V[7] * z;
        float p2 = V[2] * x + V[5] * y + V[8] * z;
        dst[r * 3 + 0] = (half_t)(s0 * p0);
        dst[r * 3 + 1] = (half_t)(s1 * p1);
        dst[r * 3 + 2] = (half_t)(s2 * p2);
    }
    dst[126] = (half_t)0.f; dst[127] = (half_t)0.f;
}

// ---------------- maxpool over N of out5 slice, broadcast to cols [0,2048) ---
__global__ __launch_bounds__(256) void maxpool_kernel(half_t* __restrict__ concat) {
    int g = blockIdx.x;                         // b*8+o
    int c = blockIdx.y * 256 + threadIdx.x;     // 0..2047
    size_t base = (size_t)g * NPTS * CCW;
    float m = -3.0e38f;
    for (int n = 0; n < NPTS; ++n)
        m = fmaxf(m, (float)concat[base + (size_t)n * CCW + 3200 + c]);
    half_t hm = (half_t)m;
    for (int n = 0; n < NPTS; ++n)
        concat[base + (size_t)n * CCW + c] = hm;
}

// ---------------- final: frame contraction, mean over o, normalize -----------
__global__ __launch_bounds__(256) void final_kernel(const float* __restrict__ Y4,
        const float* __restrict__ Vg, float* __restrict__ outNet,
        float* __restrict__ outPred) {
    int idx = blockIdx.x * 256 + threadIdx.x;  // over B*N
    if (idx >= NBATCH * NPTS) return;
    int n = idx % NPTS, b = idx / NPTS;
    float g0 = 0.f, g1 = 0.f, g2 = 0.f;
    for (int o = 0; o < 8; ++o) {
        size_t base = (((size_t)b * 8 + o) * NPTS + n) * 64;  // s4 Y ld = 64
        g0 += sgn_o(o, 0) * Y4[base + 0];
        g1 += sgn_o(o, 1) * Y4[base + 1];
        g2 += sgn_o(o, 2) * Y4[base + 2];
    }
    g0 *= 0.125f; g1 *= 0.125f; g2 *= 0.125f;
    const float* V = Vg + b * 9;
    float net[3];
    for (int i = 0; i < 3; ++i)
        net[i] = V[i * 3] * g0 + V[i * 3 + 1] * g1 + V[i * 3 + 2] * g2;
    float nn = sqrtf(net[0] * net[0] + net[1] * net[1] + net[2] * net[2]);
    float inv = 1.0f / fmaxf(nn, 1e-12f);
    for (int i = 0; i < 3; ++i) {
        outNet[idx * 3 + i] = net[i];
        outPred[idx * 3 + i] = net[i] * inv;
    }
}

// =============================== host side ===================================
extern "C" void kernel_launch(void* const* d_in, const int* in_sizes, int n_in,
                              void* d_out, int out_size, void* d_ws, size_t ws_size,
                              hipStream_t stream) {
    (void)in_sizes; (void)n_in; (void)out_size; (void)ws_size;
    const float* pcloud = (const float*)d_in[0];

    static const int dins[12]  = {60, 63, 126, 126, 128, 256, 256, 512, 5248, 256, 256, 128};
    static const int douts[12] = {63, 126, 126, 128, 256, 256, 512, 2048, 256, 256, 128, 3};
    static const int kpads[12] = {64, 64, 128, 128, 128, 256, 256, 512, 5248, 256, 256, 128};
    static const int npads[12] = {64, 128, 128, 128, 256, 256, 512, 2048, 256, 256, 128, 64};
    const float *Wp[12], *Bp[12], *Gp[12], *Bep[12];
    for (int li = 0; li < 12; ++li) {
        int base = 2 + 4 * li;
        Wp[li] = (const float*)d_in[base];
        Bp[li] = (const float*)d_in[base + 1];
        if (li < 11) {
            Gp[li]  = (const float*)d_in[base + 2];
            Bep[li] = (const float*)d_in[base + 3];
        } else { Gp[li] = 0; Bep[li] = 0; }
    }

    // ----- workspace layout (bump allocator, 256B aligned) -----
    char* ws = (char*)d_ws;
    size_t off = 0;
    auto alloc = [&](size_t bytes) -> size_t {
        off = (off + 255) & ~(size_t)255;
        size_t o = off; off += bytes; return o;
    };
    size_t wtoff[12];
    for (int li = 0; li < 12; ++li)
        wtoff[li] = alloc((size_t)npads[li] * kpads[li] * sizeof(half_t));
    size_t framed_o = alloc((size_t)MROWS * 64 * sizeof(half_t));
    size_t actA_o   = alloc((size_t)MROWS * 256 * sizeof(half_t));
    size_t actB_o   = alloc((size_t)MROWS * 256 * sizeof(half_t));
    size_t h3_o     = alloc((size_t)MROWS * 128 * sizeof(float));
    size_t eqf_o    = alloc((size_t)NBATCH * NPTS * 126 * sizeof(float));
    size_t Vl_o     = alloc((size_t)NBATCH * NPTS * 9 * sizeof(float));
    size_t cg_o     = alloc(256);
    size_t Vg_o     = alloc(256);
    size_t fr2_o    = alloc((size_t)MROWS * 128 * sizeof(half_t));
    size_t cc_o     = alloc((size_t)MROWS * CCW * sizeof(half_t));
    size_t Y_o      = alloc((size_t)MROWS * 2048 * sizeof(float));
    size_t st_o     = alloc(2 * 2048 * sizeof(float));

    half_t* framed = (half_t*)(ws + framed_o);
    half_t* actA   = (half_t*)(ws + actA_o);
    half_t* actB   = (half_t*)(ws + actB_o);
    float*  h3f    = (float*)(ws + h3_o);
    float*  eqf    = (float*)(ws + eqf_o);
    float*  Vl     = (float*)(ws + Vl_o);
    float*  c2g    = (float*)(ws + cg_o);
    float*  Vg     = (float*)(ws + Vg_o);
    half_t* fr2    = (half_t*)(ws + fr2_o);
    half_t* concat = (half_t*)(ws + cc_o);
    float*  Ybuf   = (float*)(ws + Y_o);
    float*  stats  = (float*)(ws + st_o);

    float* out_eqf  = (float*)d_out;
    float* out_net  = out_eqf + (size_t)NBATCH * NPTS * 126;
    float* out_pred = out_net + (size_t)NBATCH * NPTS * 3;

    // ----- weight prep -----
    for (int li = 0; li < 12; ++li) {
        int tot = npads[li] * kpads[li];
        prep_w_kernel<<<(tot + 255) / 256, 256, 0, stream>>>(
            Wp[li], dins[li], douts[li], (half_t*)(ws + wtoff[li]),
            kpads[li], npads[li]);
    }

    // ----- KNN + local frames + framed -----
    knn_frame_kernel<<<8, 256, 0, stream>>>(pcloud, framed, Vl);

    // ----- GEMM + BN layer runner -----
    auto runLayer = [&](int li, const half_t* Aptr, int lda, half_t* dst16,
                        int ld16, float* dst32, int ld32, int act) {
        half_t* wt = (half_t*)(ws + wtoff[li]);
        dim3 gg(npads[li] / 64, MROWS / 256);
        gemm_kernel<<<gg, 256, 0, stream>>>(Aptr, lda, wt, kpads[li], Bp[li],
                                            douts[li], Ybuf, npads[li],
                                            kpads[li], npads[li]);
        stats_kernel<<<douts[li], 256, 0, stream>>>(Ybuf, npads[li], stats);
        int tot = MROWS * npads[li];
        bn_apply_kernel<<<(tot + 255) / 256, 256, 0, stream>>>(
            Ybuf, npads[li], stats, Gp[li], Bep[li], douts[li], npads[li],
            dst16, ld16, dst32, ld32, 1.0f / (float)MROWS, act);
    };

    // e-stage
    runLayer(0, framed, 64, actA, 64, (float*)0, 0, 1);     // e1
    runLayer(1, actA, 64, actB, 128, (float*)0, 0, 1);      // e2
    runLayer(2, actB, 128, (half_t*)0, 0, h3f, 128, 0);     // e3 (fp32 out)

    // equiv_f + global frame + framed2
    {
        int tot = NBATCH * NPTS * 42;
        equivf_kernel<<<(tot + 255) / 256, 256, 0, stream>>>(h3f, Vl, eqf, out_eqf);
    }
    gframe_kernel<<<NBATCH, 256, 0, stream>>>(eqf, c2g, Vg);
    framed2_kernel<<<MROWS / 256, 256, 0, stream>>>(eqf, c2g, Vg, fr2);

    // c-stage -> concat slices: [0:2048)=max | 2048 out1 | 2176 out2 | 2432 out3
    //                           | 2688 out4 | 3200 out5
    runLayer(3, fr2, 128,           concat + 2048, CCW, (float*)0, 0, 1);  // c1
    runLayer(4, concat + 2048, CCW, concat + 2176, CCW, (float*)0, 0, 1);  // c2
    runLayer(5, concat + 2176, CCW, concat + 2432, CCW, (float*)0, 0, 1);  // c3
    runLayer(6, concat + 2432, CCW, concat + 2688, CCW, (float*)0, 0, 1);  // c4
    runLayer(7, concat + 2688, CCW, concat + 3200, CCW, (float*)0, 0, 0);  // c5

    maxpool_kernel<<<dim3(NBATCH * 8, 8), 256, 0, stream>>>(concat);

    // s-stage
    runLayer(8, concat, CCW, actA, 256, (float*)0, 0, 1);   // s1
    runLayer(9, actA, 256, actB, 256, (float*)0, 0, 1);     // s2
    runLayer(10, actB, 256, actA, 128, (float*)0, 0, 1);    // s3

    // s4: plain linear (no BN/act), fp32 out into Ybuf (ld 64)
    {
        half_t* wt = (half_t*)(ws + wtoff[11]);
        dim3 gg(1, MROWS / 256);
        gemm_kernel<<<gg, 256, 0, stream>>>(actA, 128, wt, 128, Bp[11], 3,
                                            Ybuf, 64, 128, 64);
    }

    final_kernel<<<(NBATCH * NPTS + 255) / 256, 256, 0, stream>>>(Ybuf, Vg,
                                                                  out_net, out_pred);
}
